// FPMamba_48919677501475
// MI455X (gfx1250) — matrix operand order repaired and checked by reading, not compiled, not run on hardware
//
#include <hip/hip_runtime.h>

// ---------------- problem constants ----------------
#define DM    1024          // D_MODEL
#define DI    2048          // D_INNER
#define NST   16            // D_STATE
#define DTR   64            // DT_RANK
#define LSEQ  2048
#define BSZ   2
#define MTOK  (BSZ * LSEQ)  // 4096 tokens
#define NCH   16            // scan chunks
#define CL    (LSEQ / NCH)  // 128 steps per chunk

// ---------------- WMMA types ----------------
typedef __attribute__((ext_vector_type(16))) __bf16 v16bf;
typedef __attribute__((ext_vector_type(8)))  float  v8f;

__device__ __forceinline__ __bf16 f2bf(float f) {
  union { float f; unsigned u; } cv; cv.f = f;
  unsigned u = cv.u + 0x7FFFu + ((cv.u >> 16) & 1u);   // round-to-nearest-even
  union { unsigned short s; __bf16 b; } r; r.s = (unsigned short)(u >> 16);
  return r.b;
}

union Frag { v16bf v; uint4 q[2]; };

__device__ __forceinline__ v8f wmma_bf16(v16bf a, v16bf b, v8f c) {
  return __builtin_amdgcn_wmma_f32_16x16x32_bf16(false, a, false, b, (short)0, c,
                                                 false, false);
}

// ---- gfx1250 async global->LDS copy (ASYNCcnt path) ----
__device__ __forceinline__ void async_copy_b128(unsigned lds_off, const void* gaddr) {
  asm volatile("global_load_async_to_lds_b128 %0, %1, off"
               :: "v"(lds_off), "v"(gaddr) : "memory");
}
__device__ __forceinline__ void async_wait0() {
  asm volatile("s_wait_asynccnt 0" ::: "memory");
}

// ---------------- GEMM: C(M,N) = A(M,K)bf16 @ W(N,K)bf16^T  (+epilogue) ----
// epi: 0 = none, 1 = C += addm (same shape), 2 = softplus(C + bias[col])
__device__ __forceinline__ void storeTile(v8f acc, float* __restrict__ C,
                                          const float* __restrict__ addm,
                                          const float* __restrict__ bias,
                                          int N, int row0, int col, int lh, int epi)
{
  if (col >= N) return;
  #pragma unroll
  for (int r = 0; r < 8; ++r) {
    long idx = (long)(row0 + lh * 8 + r) * N + col;
    float v = acc[r];
    if (epi == 1)       v += addm[idx];
    else if (epi == 2) { v += bias[col]; v = (v > 20.f) ? v : log1pf(__expf(v)); }
    C[idx] = v;
  }
}

__global__ __launch_bounds__(256) void k_gemm(
    const __bf16* __restrict__ A, const __bf16* __restrict__ Wt,
    float* __restrict__ C, const float* __restrict__ addm,
    const float* __restrict__ bias, int M, int N, int K, int epi)
{
  const int lane = threadIdx.x & 31;
  const int w    = threadIdx.x >> 5;           // 8 waves
  const int wm   = w >> 2, wn = w & 3;         // 2 x 4 wave grid
  const int rowBase = blockIdx.y * 64  + wm * 32;
  const int colBase = blockIdx.x * 128 + wn * 32;
  if (colBase >= N) return;                    // fully idle wave (no barriers here)
  const int l15 = lane & 15, lh = lane >> 4;

  // A fragment: lane l<16 -> row M=l, K {0..7,16..23}; l>=16 -> K {8..15,24..31}
  const long ar0 = (long)(rowBase + l15) * K + lh * 8;
  const long ar1 = ar0 + 16L * K;
  // B fragment: lane l<16 -> col N=l, K {0..7,16..23}; l>=16 -> K {8..15,24..31}
  int c0 = colBase + l15;      if (c0 > N - 1) c0 = N - 1;
  int c1 = colBase + 16 + l15; if (c1 > N - 1) c1 = N - 1;
  const long br0 = (long)c0 * K + lh * 8;
  const long br1 = (long)c1 * K + lh * 8;

  v8f acc00 = {}; v8f acc01 = {}; v8f acc10 = {}; v8f acc11 = {};

  for (int kb = 0; kb < K; kb += 32) {
    Frag a0, a1, b0, b1;
    const __bf16* pa0 = A  + ar0 + kb;
    const __bf16* pa1 = A  + ar1 + kb;
    const __bf16* pb0 = Wt + br0 + kb;
    const __bf16* pb1 = Wt + br1 + kb;
    a0.q[0] = *(const uint4*)(pa0);  a0.q[1] = *(const uint4*)(pa0 + 16);
    a1.q[0] = *(const uint4*)(pa1);  a1.q[1] = *(const uint4*)(pa1 + 16);
    b0.q[0] = *(const uint4*)(pb0);  b0.q[1] = *(const uint4*)(pb0 + 16);
    b1.q[0] = *(const uint4*)(pb1);  b1.q[1] = *(const uint4*)(pb1 + 16);
    if (kb + 288 <= K) {                       // long-range prefetch, keep near
      __builtin_prefetch(pa0 + 256, 0, 3);
      __builtin_prefetch(pb0 + 256, 0, 3);
    }
    acc00 = wmma_bf16(a0.v, b0.v, acc00);
    acc01 = wmma_bf16(a0.v, b1.v, acc01);
    acc10 = wmma_bf16(a1.v, b0.v, acc10);
    acc11 = wmma_bf16(a1.v, b1.v, acc11);
  }

  const int cs0 = colBase + l15;
  const int cs1 = colBase + 16 + l15;
  storeTile(acc00, C, addm, bias, N, rowBase,      cs0, lh, epi);
  storeTile(acc01, C, addm, bias, N, rowBase,      cs1, lh, epi);
  storeTile(acc10, C, addm, bias, N, rowBase + 16, cs0, lh, epi);
  storeTile(acc11, C, addm, bias, N, rowBase + 16, cs1, lh, epi);
}

// ---------------- elementwise helpers ----------------
__global__ __launch_bounds__(256) void k_cvt(const float* __restrict__ s,
                                             __bf16* __restrict__ d, long n)
{
  long i = (long)blockIdx.x * blockDim.x + threadIdx.x;
  long st = (long)gridDim.x * blockDim.x;
  for (; i < n; i += st) d[i] = f2bf(s[i]);
}

__global__ __launch_bounds__(256) void k_aneg(const float* __restrict__ alog,
                                              float* __restrict__ an, int n)
{
  int i = blockIdx.x * blockDim.x + threadIdx.x;
  if (i < n) an[i] = -__expf(alog[i]);
}

// split x_dbl(M,96) -> dt_low bf16 (M,64), B0 (M,16), C0 (M,16)
__global__ __launch_bounds__(256) void k_split(const float* __restrict__ xdbl,
                                               __bf16* __restrict__ dtl,
                                               float* __restrict__ B0,
                                               float* __restrict__ C0, long n)
{
  long i = (long)blockIdx.x * blockDim.x + threadIdx.x;
  long st = (long)gridDim.x * blockDim.x;
  for (; i < n; i += st) {
    long m = i / 96; int j = (int)(i % 96);
    float v = xdbl[i];
    if (j < 64)       dtl[m * 64 + j] = f2bf(v);
    else if (j < 80)  B0[m * 16 + (j - 64)] = v;
    else              C0[m * 16 + (j - 80)] = v;
  }
}

// xmy = bf16(x - y); ysh = bf16(shift1(y))
__global__ __launch_bounds__(256) void k_prep(const float* __restrict__ x,
                                              const float* __restrict__ y,
                                              __bf16* __restrict__ xmy,
                                              __bf16* __restrict__ ysh, long n)
{
  long i = (long)blockIdx.x * blockDim.x + threadIdx.x;
  long st = (long)gridDim.x * blockDim.x;
  for (; i < n; i += st) {
    long m = i / DI;
    int  t = (int)(m % LSEQ);
    xmy[i] = f2bf(x[i] - y[i]);
    ysh[i] = (t == 0) ? f2bf(0.f) : f2bf(y[i - DI]);
  }
}

// Bt = normalize(B0 + bc[:, :16]); Ct = normalize(C0 + bc[:, 16:])
__global__ __launch_bounds__(256) void k_bcnorm(const float* __restrict__ B0,
                                                const float* __restrict__ C0,
                                                const float* __restrict__ bc,
                                                float* __restrict__ Bt,
                                                float* __restrict__ Ct, int Mt)
{
  int m = blockIdx.x * blockDim.x + threadIdx.x;
  if (m >= Mt) return;
  float vb[NST], vc[NST], nb = 0.f, nc = 0.f;
  #pragma unroll
  for (int j = 0; j < NST; ++j) {
    vb[j] = B0[m * NST + j] + bc[m * 32 + j];       nb += vb[j] * vb[j];
    vc[j] = C0[m * NST + j] + bc[m * 32 + 16 + j];  nc += vc[j] * vc[j];
  }
  nb = 1.f / fmaxf(sqrtf(nb), 1e-5f);
  nc = 1.f / fmaxf(sqrtf(nc), 1e-5f);
  #pragma unroll
  for (int j = 0; j < NST; ++j) {
    Bt[m * NST + j] = vb[j] * nb;
    Ct[m * NST + j] = vc[j] * nc;
  }
}

// ---------------- chunked selective scan (3 passes, 16x parallel) ----------
// Pass 1: per-chunk local scan from h=0; record end state + cumulative decay P.
__global__ __launch_bounds__(128) void k_scan1(const float* __restrict__ u,
                                               const float* __restrict__ delta,
                                               const float* __restrict__ An,
                                               const float* __restrict__ Bt,
                                               const float* __restrict__ Ct,
                                               const float* __restrict__ Dv,
                                               float* __restrict__ y,
                                               float* __restrict__ hend,
                                               float* __restrict__ pend)
{
  const int bc = blockIdx.y;                 // b*NCH + c
  const int b  = bc >> 4, c = bc & (NCH - 1);
  const int d  = blockIdx.x * 128 + threadIdx.x;
  const int t0 = c * CL;

  __shared__ __align__(16) float sB[CL][NST];
  __shared__ __align__(16) float sC[CL][NST];
  {
    const float* gB = Bt + ((long)b * LSEQ + t0) * NST;   // contiguous 8KB
    const float* gC = Ct + ((long)b * LSEQ + t0) * NST;
    unsigned ob = (unsigned)(size_t)(&sB[0][0]);
    unsigned oc = (unsigned)(size_t)(&sC[0][0]);
    for (int k = threadIdx.x; k < CL * NST / 4; k += 128) {
      async_copy_b128(ob + k * 16, gB + k * 4);
      async_copy_b128(oc + k * 16, gC + k * 4);
    }
    async_wait0();
    __syncthreads();
  }

  float Ad[NST], h[NST], p[NST];
  #pragma unroll
  for (int j = 0; j < NST; ++j) { Ad[j] = An[d * NST + j]; h[j] = 0.f; p[j] = 1.f; }
  const float Dd = Dv[d];

  for (int tt = 0; tt < CL; ++tt) {
    long idx = ((long)b * LSEQ + t0 + tt) * DI + d;
    float dt = delta[idx];
    float uu = u[idx];
    float du = dt * uu;
    float yv = Dd * uu;
    #pragma unroll
    for (int j = 0; j < NST; ++j) {
      float e = __expf(dt * Ad[j]);
      h[j] = e * h[j] + du * sB[tt][j];
      p[j] *= e;
      yv  += h[j] * sC[tt][j];
    }
    y[idx] = yv;
  }
  long hb = ((long)bc * DI + d) * NST;
  #pragma unroll
  for (int j = 0; j < NST; ++j) { hend[hb + j] = h[j]; pend[hb + j] = p[j]; }
}

// Pass 1.5: sequentially combine the 16 chunk boundary states per channel.
__global__ __launch_bounds__(256) void k_scanfix(const float* __restrict__ hend,
                                                 const float* __restrict__ pend,
                                                 float* __restrict__ H0, int n)
{
  int i = blockIdx.x * blockDim.x + threadIdx.x;   // i -> (b, d*NST+j)
  if (i >= n) return;
  int b  = i / (DI * NST);
  int dj = i % (DI * NST);
  float h = 0.f;
  for (int c = 0; c < NCH; ++c) {
    long o = (long)(b * NCH + c) * DI * NST + dj;
    H0[o] = h;
    h = pend[o] * h + hend[o];
  }
}

// Pass 2: add cross-chunk correction  y_t += C_t . (P_t (.) h0)
__global__ __launch_bounds__(128) void k_scan2(const float* __restrict__ delta,
                                               const float* __restrict__ An,
                                               const float* __restrict__ Ct,
                                               const float* __restrict__ H0,
                                               float* __restrict__ y)
{
  const int bc = blockIdx.y;
  const int c  = bc & (NCH - 1);
  if (c == 0) return;                        // h0 == 0, nothing to add
  const int b  = bc >> 4;
  const int d  = blockIdx.x * 128 + threadIdx.x;
  const int t0 = c * CL;

  __shared__ __align__(16) float sC[CL][NST];
  {
    const float* gC = Ct + ((long)b * LSEQ + t0) * NST;
    unsigned oc = (unsigned)(size_t)(&sC[0][0]);
    for (int k = threadIdx.x; k < CL * NST / 4; k += 128)
      async_copy_b128(oc + k * 16, gC + k * 4);
    async_wait0();
    __syncthreads();
  }

  float Ad[NST], ph[NST];
  #pragma unroll
  for (int j = 0; j < NST; ++j) {
    Ad[j] = An[d * NST + j];
    ph[j] = H0[((long)bc * DI + d) * NST + j];   // P_t * h0, P starts at 1
  }
  for (int tt = 0; tt < CL; ++tt) {
    long idx = ((long)b * LSEQ + t0 + tt) * DI + d;
    float dt = delta[idx];
    float yv = 0.f;
    #pragma unroll
    for (int j = 0; j < NST; ++j) {
      ph[j] *= __expf(dt * Ad[j]);
      yv   += ph[j] * sC[tt][j];
    }
    y[idx] += yv;
  }
}

// LayerNorm of yt row + fixed-point update of y (relaxed or final)
__global__ __launch_bounds__(256) void k_ln(const float* __restrict__ yt,
                                            const float* __restrict__ gamma,
                                            const float* __restrict__ beta,
                                            float* __restrict__ y, int finalFlag)
{
  const int m = blockIdx.x;
  const float* row = yt + (long)m * DI;
  float s = 0.f, ss = 0.f;
  for (int i = threadIdx.x; i < DI; i += 256) { float v = row[i]; s += v; ss += v * v; }
  __shared__ float rs[256], rq[256];
  rs[threadIdx.x] = s; rq[threadIdx.x] = ss;
  __syncthreads();
  for (int st = 128; st > 0; st >>= 1) {
    if (threadIdx.x < st) { rs[threadIdx.x] += rs[threadIdx.x + st];
                            rq[threadIdx.x] += rq[threadIdx.x + st]; }
    __syncthreads();
  }
  const float mean = rs[0] * (1.f / DI);
  const float var  = rq[0] * (1.f / DI) - mean * mean;
  const float inv  = rsqrtf(var + 1e-5f);
  float* yrow = y + (long)m * DI;
  for (int i = threadIdx.x; i < DI; i += 256) {
    float ln = gamma[i] * (row[i] - mean) * inv + beta[i];
    if (finalFlag) yrow[i] = ln;
    else { float p = yrow[i]; yrow[i] = p + 0.99f * (ln - p); }
  }
}

// g = bf16( silu(z2) * y )
__global__ __launch_bounds__(256) void k_gate(const float* __restrict__ z2,
                                              const float* __restrict__ y,
                                              __bf16* __restrict__ g, long n)
{
  long i = (long)blockIdx.x * blockDim.x + threadIdx.x;
  long st = (long)gridDim.x * blockDim.x;
  for (; i < n; i += st) {
    float z = z2[i];
    float sg = z / (1.f + __expf(-z));
    g[i] = f2bf(sg * y[i]);
  }
}

// ---------------- host orchestration ----------------
extern "C" void kernel_launch(void* const* d_in, const int* in_sizes, int n_in,
                              void* d_out, int out_size, void* d_ws, size_t ws_size,
                              hipStream_t stream)
{
  const float* hs    = (const float*)d_in[0];
  const float* W_in  = (const float*)d_in[1];
  const float* W_x   = (const float*)d_in[2];
  const float* W_dt  = (const float*)d_in[3];
  const float* b_dt  = (const float*)d_in[4];
  const float* A_log = (const float*)d_in[5];
  const float* W_bc  = (const float*)d_in[6];
  const float* W_zy  = (const float*)d_in[7];
  const float* W_mix = (const float*)d_in[8];
  const float* Dvec  = (const float*)d_in[9];
  const float* ln_g  = (const float*)d_in[10];
  const float* ln_b  = (const float*)d_in[11];
  const float* W_out = (const float*)d_in[12];
  float* out = (float*)d_out;

  char* ws = (char*)d_ws;
  size_t off = 0;
  auto alloc = [&](size_t bytes) -> char* {
    char* p = ws + off;
    off += (bytes + 255) & ~(size_t)255;
    return p;
  };

  const size_t MD = (size_t)MTOK * DI;           // 8M elems
  const size_t HB = (size_t)BSZ * NCH * DI * NST; // chunk state elems (1M)
  float* f_x    = (float*)alloc(MD * 4);
  float* f_z    = (float*)alloc(MD * 4);
  float* f_dl   = (float*)alloc(MD * 4);         // delta
  float* f_y    = (float*)alloc(MD * 4);
  float* f_yt   = (float*)alloc(MD * 4);         // scan output / LN input
  float* f_xt   = (float*)alloc(MD * 4);         // x_tilde, reused as z2
  float* f_B0   = (float*)alloc((size_t)MTOK * NST * 4);
  float* f_C0   = (float*)alloc((size_t)MTOK * NST * 4);
  float* f_Bt   = (float*)alloc((size_t)MTOK * NST * 4);
  float* f_Ct   = (float*)alloc((size_t)MTOK * NST * 4);
  float* f_xdbl = (float*)alloc((size_t)MTOK * 96 * 4);
  float* f_bc   = (float*)alloc((size_t)MTOK * 32 * 4);
  float* f_A    = (float*)alloc((size_t)DI * NST * 4);
  float* f_hend = (float*)alloc(HB * 4);
  float* f_pend = (float*)alloc(HB * 4);
  float* f_H0   = (float*)alloc(HB * 4);

  __bf16* b_r1   = (__bf16*)alloc(MD * 2);       // x_bf / xmy_bf / g_bf
  __bf16* b_r2   = (__bf16*)alloc(MD * 2);       // hs_bf / ysh_bf
  __bf16* b_dtl  = (__bf16*)alloc((size_t)MTOK * DTR * 2);
  __bf16* b_Win  = (__bf16*)alloc((size_t)2 * DI * DM * 2);
  __bf16* b_Wx   = (__bf16*)alloc((size_t)96 * DI * 2);
  __bf16* b_Wdt  = (__bf16*)alloc((size_t)DI * DTR * 2);
  __bf16* b_Wbc  = (__bf16*)alloc((size_t)32 * DI * 2);
  __bf16* b_Wzy  = (__bf16*)alloc((size_t)DI * DI * 2);
  __bf16* b_Wmix = (__bf16*)alloc((size_t)DI * DI * 2);
  __bf16* b_Wout = (__bf16*)alloc((size_t)DM * DI * 2);
  (void)ws_size; (void)in_sizes; (void)n_in; (void)out_size;

  auto cvt = [&](const float* s, __bf16* d, size_t n) {
    k_cvt<<<dim3(1024), dim3(256), 0, stream>>>(s, d, (long)n);
  };
  auto gemm = [&](const __bf16* A, const __bf16* Wt, float* C, int Mm, int Nn, int Kk,
                  const float* addm, const float* bias, int epi) {
    dim3 g((Nn + 127) / 128, (Mm + 63) / 64);
    k_gemm<<<g, dim3(256), 0, stream>>>(A, Wt, C, addm, bias, Mm, Nn, Kk, epi);
  };

  // ---- one-time precompute ----
  cvt(hs, b_r2, (size_t)MTOK * DM);
  cvt(W_in,  b_Win,  (size_t)2 * DI * DM);
  cvt(W_x,   b_Wx,   (size_t)96 * DI);
  cvt(W_dt,  b_Wdt,  (size_t)DI * DTR);
  cvt(W_bc,  b_Wbc,  (size_t)32 * DI);
  cvt(W_zy,  b_Wzy,  (size_t)DI * DI);
  cvt(W_mix, b_Wmix, (size_t)DI * DI);
  cvt(W_out, b_Wout, (size_t)DM * DI);
  k_aneg<<<dim3((DI * NST + 255) / 256), dim3(256), 0, stream>>>(A_log, f_A, DI * NST);

  // xz = hs @ W_in^T   (x / z via W_in row halves)
  gemm(b_r2, b_Win,                   f_x, MTOK, DI, DM, nullptr, nullptr, 0);
  gemm(b_r2, b_Win + (size_t)DI * DM, f_z, MTOK, DI, DM, nullptr, nullptr, 0);
  cvt(f_x, b_r1, MD);
  gemm(b_r1, b_Wx, f_xdbl, MTOK, 96, DI, nullptr, nullptr, 0);
  k_split<<<dim3(1024), dim3(256), 0, stream>>>(f_xdbl, b_dtl, f_B0, f_C0,
                                                (long)MTOK * 96);
  gemm(b_dtl, b_Wdt, f_dl, MTOK, DI, DTR, nullptr, b_dt, 2);

  hipMemsetAsync(f_y, 0, MD * 4, stream);        // y0 = 0

  // ---- fixed-point iterations: 8 relaxed + 1 final ----
  for (int it = 0; it < 9; ++it) {
    k_prep<<<dim3(1024), dim3(256), 0, stream>>>(f_x, f_y, b_r1, b_r2, (long)MD);
    gemm(b_r1, b_Wmix, f_xt, MTOK, DI, DI, f_y, nullptr, 1);   // x_tilde
    gemm(b_r2, b_Wbc,  f_bc, MTOK, 32, DI, nullptr, nullptr, 0);
    k_bcnorm<<<dim3((MTOK + 255) / 256), dim3(256), 0, stream>>>(f_B0, f_C0, f_bc,
                                                                 f_Bt, f_Ct, MTOK);
    k_scan1<<<dim3(DI / 128, BSZ * NCH), dim3(128), 0, stream>>>(
        f_xt, f_dl, f_A, f_Bt, f_Ct, Dvec, f_yt, f_hend, f_pend);
    k_scanfix<<<dim3((BSZ * DI * NST + 255) / 256), dim3(256), 0, stream>>>(
        f_hend, f_pend, f_H0, BSZ * DI * NST);
    k_scan2<<<dim3(DI / 128, BSZ * NCH), dim3(128), 0, stream>>>(
        f_dl, f_A, f_Ct, f_H0, f_yt);
    k_ln<<<dim3(MTOK), dim3(256), 0, stream>>>(f_yt, ln_g, ln_b, f_y,
                                               (it == 8) ? 1 : 0);
  }

  // ---- output head ----
  k_prep<<<dim3(1024), dim3(256), 0, stream>>>(f_x, f_y, b_r1, b_r2, (long)MD);
  gemm(b_r2, b_Wzy, f_xt, MTOK, DI, DI, f_z, nullptr, 1);      // z2
  k_gate<<<dim3(1024), dim3(256), 0, stream>>>(f_xt, f_y, b_r1, (long)MD);
  gemm(b_r1, b_Wout, out, MTOK, DM, DI, nullptr, nullptr, 0);  // output
}
